// ConvBnA_int_12352325943970
// MI455X (gfx1250) — 8-bit matrix operand order repaired and checked by reading, compiled, not run on hardware
//
#include <hip/hip_runtime.h>

typedef int v8i __attribute__((ext_vector_type(8)));
typedef int v4i __attribute__((ext_vector_type(4)));

#define HPIX   56
#define WPIX   56
#define HP     58
#define CIN    128
#define COUT   256
#define BATCH  32
#define HWPIX  (HPIX * WPIX)
#define NPIX   (BATCH * HWPIX)
#define KCHUNKS 18
#define PADPIX (BATCH * HP * HP)
#define XPAD_BYTES ((size_t)PADPIX * CIN)
#define WFRAG_DWORDS (16 * KCHUNKS * 32 * 8)

__global__ __launch_bounds__(256) void pack_x_kernel(const int* __restrict__ x,
                                                     signed char* __restrict__ xp) {
  int pix = blockIdx.x * 256 + threadIdx.x;
  if (pix >= PADPIX) return;
  int b  = pix / (HP * HP);
  int r  = pix - b * (HP * HP);
  int hp = r / HP;
  int wp = r - hp * HP;
  unsigned* dst = (unsigned*)(xp + (size_t)pix * CIN);
  if (hp >= 1 && hp <= HPIX && wp >= 1 && wp <= WPIX) {
    int hw = (hp - 1) * WPIX + (wp - 1);
    const int* src = x + (size_t)b * CIN * HWPIX + hw;
#pragma unroll 4
    for (int c4 = 0; c4 < CIN / 4; ++c4) {
      unsigned v0 = (unsigned)src[(4 * c4 + 0) * HWPIX] & 0xFFu;
      unsigned v1 = (unsigned)src[(4 * c4 + 1) * HWPIX] & 0xFFu;
      unsigned v2 = (unsigned)src[(4 * c4 + 2) * HWPIX] & 0xFFu;
      unsigned v3 = (unsigned)src[(4 * c4 + 3) * HWPIX] & 0xFFu;
      dst[c4] = v0 | (v1 << 8) | (v2 << 16) | (v3 << 24);
    }
  } else {
#pragma unroll 8
    for (int c4 = 0; c4 < CIN / 4; ++c4) dst[c4] = 0u;
  }
}

__global__ __launch_bounds__(256) void pack_w_kernel(const int* __restrict__ wt,
                                                     unsigned* __restrict__ wf) {
  int idx = blockIdx.x * 256 + threadIdx.x;
  if (idx >= WFRAG_DWORDS) return;
  int v     = idx & 7;
  int lane  = (idx >> 3) & 31;
  int kc    = (idx >> 8) % KCHUNKS;
  int mtile = (idx >> 8) / KCHUNKS;
  int m     = mtile * 16 + (lane & 15);
  int hi    = lane >> 4;
  int kbase = kc * 64 + (v >> 2) * 32 + hi * 16 + (v & 3) * 4;
  unsigned packed = 0;
#pragma unroll
  for (int j = 0; j < 4; ++j) {
    int k   = kbase + j;
    int pos = k >> 7;
    int c   = k & 127;
    int val = wt[((size_t)m * CIN + c) * 9 + pos];
    packed |= ((unsigned)val & 0xFFu) << (8 * j);
  }
  wf[idx] = packed;
}

__global__ __launch_bounds__(256) void conv_gemm_kernel(
    const signed char* __restrict__ xp, const signed char* __restrict__ wf,
    const int* __restrict__ tq, const int* __restrict__ nq,
    const int* __restrict__ aminq, const int* __restrict__ amaxq,
    float* __restrict__ out) {
  const int lane  = threadIdx.x & 31;
  const int wave  = threadIdx.x >> 5;
  const int mtile = blockIdx.y * 8 + wave;
  const int nbase = blockIdx.x * 64;
  const int col   = lane & 15;
  const int hi    = lane >> 4;

  int p[4];
  const signed char* bptr[4];
#pragma unroll
  for (int ti = 0; ti < 4; ++ti) {
    p[ti]  = nbase + ti * 16 + col;
    int b  = p[ti] / HWPIX;
    int hw = p[ti] - b * HWPIX;
    int h  = hw / WPIX;
    int w  = hw - h * WPIX;
    size_t ctr = ((size_t)(b * HP + h + 1) * HP + (w + 1)) * CIN;
    bptr[ti] = xp + ctr + hi * 16;
  }

  const signed char* a_ptr = wf + (size_t)mtile * KCHUNKS * 1024 + lane * 32;

#define CHUNK_OFF(kc) ((((kc) >> 1) / 3 - 1) * HP * CIN + \
                       (((kc) >> 1) % 3 - 1) * CIN + (((kc) & 1) << 6))

  v8i acc[4];
  const v8i zero8 = {0, 0, 0, 0, 0, 0, 0, 0};
#pragma unroll
  for (int ti = 0; ti < 4; ++ti) acc[ti] = zero8;

  v8i a_cur, b_cur[4];
  {
    v4i* ah = (v4i*)&a_cur;
    ah[0] = *(const v4i*)(a_ptr);
    ah[1] = *(const v4i*)(a_ptr + 16);
    const int off0 = CHUNK_OFF(0);
#pragma unroll
    for (int ti = 0; ti < 4; ++ti) {
      v4i* bh = (v4i*)&b_cur[ti];
      bh[0] = *(const v4i*)(bptr[ti] + off0);
      bh[1] = *(const v4i*)(bptr[ti] + off0 + 32);
    }
  }

#pragma unroll
  for (int kc = 0; kc < KCHUNKS; ++kc) {
    v8i a_nxt, b_nxt[4];
    if (kc + 1 < KCHUNKS) {
      const int offn = CHUNK_OFF(kc + 1);
      v4i* ah = (v4i*)&a_nxt;
      ah[0] = *(const v4i*)(a_ptr + (kc + 1) * 1024);
      ah[1] = *(const v4i*)(a_ptr + (kc + 1) * 1024 + 16);
#pragma unroll
      for (int ti = 0; ti < 4; ++ti) {
        v4i* bh = (v4i*)&b_nxt[ti];
        bh[0] = *(const v4i*)(bptr[ti] + offn);
        bh[1] = *(const v4i*)(bptr[ti] + offn + 32);
      }
    }
#pragma unroll
    for (int ti = 0; ti < 4; ++ti) {
      acc[ti] = __builtin_amdgcn_wmma_i32_16x16x64_iu8(
          true, a_cur, true, b_cur[ti], acc[ti], false, false);
    }
    if (kc + 1 < KCHUNKS) {
      a_cur = a_nxt;
#pragma unroll
      for (int ti = 0; ti < 4; ++ti) b_cur[ti] = b_nxt[ti];
    }
  }
#undef CHUNK_OFF

  int tv[8], sv[8], lv[8], hv[8];
#pragma unroll
  for (int r = 0; r < 8; ++r) {
    int cout = mtile * 16 + r + hi * 8;
    tv[r] = tq[cout];
    sv[r] = -nq[cout];
    lv[r] = aminq[cout];
    hv[r] = amaxq[cout];
  }

#pragma unroll
  for (int ti = 0; ti < 4; ++ti) {
    int pt = p[ti];
    int b  = pt / HWPIX;
    int hw = pt - b * HWPIX;
    float* obase = out + (size_t)b * COUT * HWPIX + hw +
                   (size_t)(mtile * 16 + hi * 8) * HWPIX;
#pragma unroll
    for (int r = 0; r < 8; ++r) {
      int val = acc[ti][r] + tv[r];
      val = val >> sv[r];
      val = val < lv[r] ? lv[r] : val;
      val = val > hv[r] ? hv[r] : val;
      obase[(size_t)r * HWPIX] = (float)val;
    }
  }
}

extern "C" void kernel_launch(void* const* d_in, const int* in_sizes, int n_in,
                              void* d_out, int out_size, void* d_ws, size_t ws_size,
                              hipStream_t stream) {
  const int* x    = (const int*)d_in[0];
  const int* wt   = (const int*)d_in[1];
  const int* tq   = (const int*)d_in[2];
  const int* nq   = (const int*)d_in[3];
  const int* amin = (const int*)d_in[4];
  const int* amax = (const int*)d_in[5];
  float* out = (float*)d_out;

  signed char* xp = (signed char*)d_ws;
  unsigned*    wf = (unsigned*)((char*)d_ws + XPAD_BYTES);

  pack_x_kernel<<<dim3((PADPIX + 255) / 256), dim3(256), 0, stream>>>(x, xp);
  pack_w_kernel<<<dim3(WFRAG_DWORDS / 256), dim3(256), 0, stream>>>(wt, wf);
  conv_gemm_kernel<<<dim3(NPIX / 64, 2), dim3(256), 0, stream>>>(
      xp, (const signed char*)wf, tq, nq, amin, amax, out);
}
